// LoRAAttention_64166811402405
// MI455X (gfx1250) — matrix operand order repaired and checked by reading, not compiled, not run on hardware
//
#include <hip/hip_runtime.h>

typedef __attribute__((ext_vector_type(16))) _Float16 v16h;
typedef __attribute__((ext_vector_type(8)))  _Float16 v8h;
typedef __attribute__((ext_vector_type(8)))  float    v8f;

#define NHEADS 12
#define HD     64
#define CDIM   768
#define NSEQ   1024
#define NBATCH 8
#define NTOK   (NBATCH * NSEQ)

// ---------------------------------------------------------------------------
// WMMA helpers (gfx1250 wave32, 16x16x32 f16 -> f32)
// ---------------------------------------------------------------------------
__device__ __forceinline__ v8f wmma16(v16h a, v16h b, v8f c) {
  return __builtin_amdgcn_wmma_f32_16x16x32_f16(false, a, false, b, (short)0, c,
                                                false, false);
}

// A-matrix fragment (16x32 f16) from row-major LDS tile [rows][stride].
__device__ __forceinline__ v16h frag_a(const _Float16* p0, int stride, int row,
                                       int kbase, int lane) {
  const int half = (lane & 16) ? 8 : 0;
  const _Float16* p = p0 + (size_t)row * stride + kbase + half;
  union { v16h v; v8h h[2]; } u;
  u.h[0] = *(const v8h*)(p);
  u.h[1] = *(const v8h*)(p + 16);
  return u.v;
}

// B-matrix fragment (32x16 f16) from LDS tile stored [col][stride] (K contiguous).
__device__ __forceinline__ v16h frag_b(const _Float16* p0, int stride, int col,
                                       int kbase, int lane) {
  const int half = (lane & 16) ? 8 : 0;
  const _Float16* p = p0 + (size_t)col * stride + kbase + half;
  union { v16h v; v8h h[2]; } u;
  u.h[0] = *(const v8h*)(p);
  u.h[1] = *(const v8h*)(p + 16);
  return u.v;
}

// ---------------------------------------------------------------------------
// CDNA5 async global->LDS copy (ASYNCcnt path).  INST_OFFSET is added to both
// the LDS and the global address (ISA 08_async_tensor §4.4), so one base pair
// plus offset immediates covers a contiguous chunk.
// ---------------------------------------------------------------------------
__device__ __forceinline__ unsigned lds_u32(const void* p) {
  // generic LDS pointer = {aperture[63:32], as3 offset[31:0]}
  return (unsigned)(unsigned long long)p;
}

template <int OFF>
__device__ __forceinline__ void async_ld128(unsigned lds, const void* g) {
  asm volatile("global_load_async_to_lds_b128 %0, %1, off offset:%2"
               :: "v"(lds), "v"(g), "i"(OFF)
               : "memory");
}

__device__ __forceinline__ void async_copy32B(unsigned lds, const void* g) {
  async_ld128<0>(lds, g);
  async_ld128<16>(lds, g);
}

__device__ __forceinline__ void async_copy64B(unsigned lds, const void* g) {
  async_ld128<0>(lds, g);
  async_ld128<16>(lds, g);
  async_ld128<32>(lds, g);
  async_ld128<48>(lds, g);
}

__device__ __forceinline__ void wait_async0() {
  asm volatile("s_wait_asynccnt 0x0" ::: "memory");
}

// ---------------------------------------------------------------------------
// Small prep kernels
// ---------------------------------------------------------------------------
__global__ void k_convert(const float* __restrict__ in, _Float16* __restrict__ out,
                          int n) {
  int i = blockIdx.x * blockDim.x + threadIdx.x;
  if (i < n) out[i] = (_Float16)in[i];
}

// M = I + A(64x4) @ B(4x64)
__global__ void k_make_m(const float* __restrict__ A, const float* __restrict__ B,
                         float* __restrict__ M) {
  int i = blockIdx.x;   // 64
  int j = threadIdx.x;  // 64
  float s = (i == j) ? 1.0f : 0.0f;
#pragma unroll
  for (int r = 0; r < 4; ++r) s += A[i * 4 + r] * B[r * 64 + j];
  M[i * 64 + j] = s;
}

// Fold LoRA into Wqkv: q-head rows <- Mq^T @ Wq_h, v-head rows <- Mv^T @ Wv_h.
__global__ __launch_bounds__(256) void k_fold_w(const float* __restrict__ W,
                                                const float* __restrict__ Mq,
                                                const float* __restrict__ Mv,
                                                _Float16* __restrict__ Wh) {
  long idx = (long)blockIdx.x * 256 + threadIdx.x;
  if (idx >= (long)3 * CDIM * CDIM) return;
  int row = (int)(idx / CDIM), c = (int)(idx % CDIM);
  int seg = row / CDIM;  // 0=q, 1=k, 2=v
  float s;
  if (seg == 1) {
    s = W[idx];
  } else {
    const float* M = (seg == 0) ? Mq : Mv;
    int i = row & 63;
    long base = (long)(row - i) * CDIM + c;
    s = 0.0f;
#pragma unroll 8
    for (int k = 0; k < 64; ++k) s += M[k * 64 + i] * W[base + (long)k * CDIM];
  }
  Wh[idx] = (_Float16)s;
}

// ---------------------------------------------------------------------------
// GEMM 1: Y = Xh(8192x768) @ Wh^T(2304x768), scatter into Q/K/V [B,H,N,64] f16
// Double-buffered LDS, async global->LDS staging.
// ---------------------------------------------------------------------------
__global__ __launch_bounds__(256) void k_gemm_qkv(const _Float16* __restrict__ Xh,
                                                  const _Float16* __restrict__ Wh,
                                                  _Float16* __restrict__ Qb,
                                                  _Float16* __restrict__ Kb,
                                                  _Float16* __restrict__ Vb) {
  __shared__ __align__(32) _Float16 sA[2][128 * 32];
  __shared__ __align__(32) _Float16 sB[2][128 * 32];
  const int tid = threadIdx.x, lane = tid & 31, wave = tid >> 5;
  const int m0 = blockIdx.x * 128;
  const int n0 = blockIdx.y * 128;
  const int wm0 = (wave & 3) * 32;   // 4 waves along M
  const int wn0 = (wave >> 2) * 64;  // 2 waves along N
  const int lr = tid >> 1, ls = (tid & 1) << 4;  // copy: row, 16-half segment
  const _Float16* gA = &Xh[(size_t)(m0 + lr) * CDIM + ls];
  const _Float16* gB = &Wh[(size_t)(n0 + lr) * CDIM + ls];
  const unsigned laA[2] = { lds_u32(&sA[0][lr * 32 + ls]), lds_u32(&sA[1][lr * 32 + ls]) };
  const unsigned laB[2] = { lds_u32(&sB[0][lr * 32 + ls]), lds_u32(&sB[1][lr * 32 + ls]) };

  // prologue: stage tile 0
  async_copy32B(laA[0], gA);
  async_copy32B(laB[0], gB);
  wait_async0();
  __syncthreads();

  v8f acc[2][4] = {};
  for (int k0 = 0; k0 < CDIM; k0 += 32) {
    const int cur = (k0 >> 5) & 1;
    if (k0 + 32 < CDIM) {  // prefetch next tile into other buffer
      async_copy32B(laA[cur ^ 1], gA + k0 + 32);
      async_copy32B(laB[cur ^ 1], gB + k0 + 32);
    }
    v16h af[2], bf[4];
#pragma unroll
    for (int i = 0; i < 2; ++i)
      af[i] = frag_a(sA[cur], 32, wm0 + i * 16 + (lane & 15), 0, lane);
#pragma unroll
    for (int j = 0; j < 4; ++j)
      bf[j] = frag_b(sB[cur], 32, wn0 + j * 16 + (lane & 15), 0, lane);
#pragma unroll
    for (int i = 0; i < 2; ++i)
#pragma unroll
      for (int j = 0; j < 4; ++j) acc[i][j] = wmma16(af[i], bf[j], acc[i][j]);
    if (k0 + 32 < CDIM) wait_async0();
    __syncthreads();
  }

  // Epilogue: scatter f16 into Q/K/V [B,H,N,hd]
#pragma unroll
  for (int i = 0; i < 2; ++i) {
    int rbase = m0 + wm0 + i * 16 + ((lane & 16) ? 8 : 0);
#pragma unroll
    for (int j = 0; j < 4; ++j) {
      int col = n0 + wn0 + j * 16 + (lane & 15);
      int seg = col / CDIM;  // 0=q, 1=k, 2=v
      int jj = col - seg * CDIM;
      int h = jj >> 6, d = jj & 63;
      _Float16* dst = (seg == 0) ? Qb : (seg == 1) ? Kb : Vb;
#pragma unroll
      for (int v = 0; v < 8; ++v) {
        int t = rbase + v;
        int b = t >> 10, n = t & 1023;
        dst[((((size_t)b * NHEADS + h) * NSEQ + n) << 6) + d] = (_Float16)acc[i][j][v];
      }
    }
  }
}

// ---------------------------------------------------------------------------
// Flash attention: one block per (b*H+h, 64-row query block). 4 waves x 16 rows.
// ---------------------------------------------------------------------------
__global__ __launch_bounds__(128) void k_flash(const _Float16* __restrict__ Qb,
                                               const _Float16* __restrict__ Kb,
                                               const _Float16* __restrict__ Vb,
                                               _Float16* __restrict__ Pin) {
  __shared__ __align__(32) _Float16 sQ[64 * 64];
  __shared__ __align__(32) _Float16 sK[64 * 64];
  __shared__ __align__(32) _Float16 sVt[64 * 64];   // transposed: [d][key]
  __shared__ __align__(32) _Float16 sP[4][16 * 64]; // per-wave P tile
  const int tid = threadIdx.x, lane = tid & 31, wave = tid >> 5;
  const int bh = blockIdx.x;        // b*NHEADS + h
  const int q0 = blockIdx.y * 64;
  const int b = bh / NHEADS, h = bh - b * NHEADS;
  const size_t base = (size_t)bh * NSEQ * HD;
  const int cr = tid >> 1, cs = (tid & 1) << 5;  // copy: row, 32-half segment
  const unsigned lK = lds_u32(&sK[cr * HD + cs]);

  // stage Q tile via async path
  async_copy64B(lds_u32(&sQ[cr * HD + cs]), &Qb[base + (size_t)(q0 + cr) * HD + cs]);
  wait_async0();
  __syncthreads();

  v16h qf[2];
  qf[0] = frag_a(sQ, HD, wave * 16 + (lane & 15), 0, lane);
  qf[1] = frag_a(sQ, HD, wave * 16 + (lane & 15), 32, lane);
  float mrow[8], lrow[8];
  v8f of[4] = {};
#pragma unroll
  for (int v = 0; v < 8; ++v) { mrow[v] = -1e30f; lrow[v] = 0.0f; }
  const float scale = 0.125f;  // 64^-0.5

  for (int kb = 0; kb < 16; ++kb) {
    __syncthreads();
    // K tile: async global->LDS
    async_copy64B(lK, &Kb[base + (size_t)(kb * 64 + cr) * HD + cs]);
    {  // V transposed into LDS: sVt[d][key] (manual, async can't transpose)
      const _Float16* g = &Vb[base + (size_t)(kb * 64 + cr) * HD + cs];
#pragma unroll 8
      for (int d = 0; d < 32; ++d) sVt[(cs + d) * 64 + cr] = g[d];
    }
    wait_async0();
    __syncthreads();
    // S = Q @ K^T (4 column fragments of 16 keys each)
    v8f sf[4];
#pragma unroll
    for (int n = 0; n < 4; ++n) {
      v8f s = {};
      s = wmma16(qf[0], frag_b(sK, HD, n * 16 + (lane & 15), 0, lane), s);
      s = wmma16(qf[1], frag_b(sK, HD, n * 16 + (lane & 15), 32, lane), s);
      sf[n] = s;
    }
#pragma unroll
    for (int n = 0; n < 4; ++n)
#pragma unroll
      for (int v = 0; v < 8; ++v) sf[n][v] *= scale;
    // online softmax: each row lives across a 16-lane column group
    float alpha[8];
#pragma unroll
    for (int v = 0; v < 8; ++v) {
      float mx = fmaxf(fmaxf(sf[0][v], sf[1][v]), fmaxf(sf[2][v], sf[3][v]));
#pragma unroll
      for (int off = 1; off < 16; off <<= 1) mx = fmaxf(mx, __shfl_xor(mx, off, 16));
      float mn = fmaxf(mrow[v], mx);
      alpha[v] = __expf(mrow[v] - mn);
      mrow[v] = mn;
    }
    float rs[8];
#pragma unroll
    for (int v = 0; v < 8; ++v) rs[v] = 0.0f;
#pragma unroll
    for (int n = 0; n < 4; ++n)
#pragma unroll
      for (int v = 0; v < 8; ++v) {
        float p = __expf(sf[n][v] - mrow[v]);
        sf[n][v] = p;
        rs[v] += p;
      }
#pragma unroll
    for (int v = 0; v < 8; ++v) {
      float s = rs[v];
#pragma unroll
      for (int off = 1; off < 16; off <<= 1) s += __shfl_xor(s, off, 16);
      lrow[v] = lrow[v] * alpha[v] + s;
    }
#pragma unroll
    for (int n = 0; n < 4; ++n)
#pragma unroll
      for (int v = 0; v < 8; ++v) of[n][v] *= alpha[v];
    // reshape P: D-fragment -> LDS -> A-fragment
    _Float16* pw = sP[wave];
    const int pr = (lane & 16) ? 8 : 0;
#pragma unroll
    for (int n = 0; n < 4; ++n)
#pragma unroll
      for (int v = 0; v < 8; ++v)
        pw[(pr + v) * 64 + n * 16 + (lane & 15)] = (_Float16)sf[n][v];
    __syncthreads();
    v16h pf0 = frag_a(pw, 64, lane & 15, 0, lane);
    v16h pf1 = frag_a(pw, 64, lane & 15, 32, lane);
#pragma unroll
    for (int n = 0; n < 4; ++n) {
      of[n] = wmma16(pf0, frag_b(sVt, 64, n * 16 + (lane & 15), 0, lane), of[n]);
      of[n] = wmma16(pf1, frag_b(sVt, 64, n * 16 + (lane & 15), 32, lane), of[n]);
    }
  }
  // normalize and write proj input [B*N][C] at column h*64 + d (f16)
  const int tr = (lane & 16) ? 8 : 0;
#pragma unroll
  for (int n = 0; n < 4; ++n)
#pragma unroll
    for (int v = 0; v < 8; ++v) {
      int r = q0 + wave * 16 + tr + v;
      int col = h * HD + n * 16 + (lane & 15);
      Pin[(size_t)(b * NSEQ + r) * CDIM + col] = (_Float16)(of[n][v] / lrow[v]);
    }
}

// ---------------------------------------------------------------------------
// GEMM 2: Out = Pin(8192x768) @ Wph^T(768x768) + bias, f32 output
// ---------------------------------------------------------------------------
__global__ __launch_bounds__(256) void k_gemm_proj(const _Float16* __restrict__ Pin,
                                                   const _Float16* __restrict__ Wph,
                                                   const float* __restrict__ bias,
                                                   float* __restrict__ Out) {
  __shared__ __align__(32) _Float16 sA[2][128 * 32];
  __shared__ __align__(32) _Float16 sB[2][128 * 32];
  const int tid = threadIdx.x, lane = tid & 31, wave = tid >> 5;
  const int m0 = blockIdx.x * 128;
  const int n0 = blockIdx.y * 128;
  const int wm0 = (wave & 3) * 32;
  const int wn0 = (wave >> 2) * 64;
  const int lr = tid >> 1, ls = (tid & 1) << 4;
  const _Float16* gA = &Pin[(size_t)(m0 + lr) * CDIM + ls];
  const _Float16* gB = &Wph[(size_t)(n0 + lr) * CDIM + ls];
  const unsigned laA[2] = { lds_u32(&sA[0][lr * 32 + ls]), lds_u32(&sA[1][lr * 32 + ls]) };
  const unsigned laB[2] = { lds_u32(&sB[0][lr * 32 + ls]), lds_u32(&sB[1][lr * 32 + ls]) };

  async_copy32B(laA[0], gA);
  async_copy32B(laB[0], gB);
  wait_async0();
  __syncthreads();

  v8f acc[2][4] = {};
  for (int k0 = 0; k0 < CDIM; k0 += 32) {
    const int cur = (k0 >> 5) & 1;
    if (k0 + 32 < CDIM) {
      async_copy32B(laA[cur ^ 1], gA + k0 + 32);
      async_copy32B(laB[cur ^ 1], gB + k0 + 32);
    }
    v16h af[2], bf[4];
#pragma unroll
    for (int i = 0; i < 2; ++i)
      af[i] = frag_a(sA[cur], 32, wm0 + i * 16 + (lane & 15), 0, lane);
#pragma unroll
    for (int j = 0; j < 4; ++j)
      bf[j] = frag_b(sB[cur], 32, wn0 + j * 16 + (lane & 15), 0, lane);
#pragma unroll
    for (int i = 0; i < 2; ++i)
#pragma unroll
      for (int j = 0; j < 4; ++j) acc[i][j] = wmma16(af[i], bf[j], acc[i][j]);
    if (k0 + 32 < CDIM) wait_async0();
    __syncthreads();
  }
#pragma unroll
  for (int i = 0; i < 2; ++i) {
    int rbase = m0 + wm0 + i * 16 + ((lane & 16) ? 8 : 0);
#pragma unroll
    for (int j = 0; j < 4; ++j) {
      int col = n0 + wn0 + j * 16 + (lane & 15);
      float bc = bias[col];
#pragma unroll
      for (int v = 0; v < 8; ++v)
        Out[(size_t)(rbase + v) * CDIM + col] = acc[i][j][v] + bc;
    }
  }
}

// ---------------------------------------------------------------------------
// Host launcher
// ---------------------------------------------------------------------------
extern "C" void kernel_launch(void* const* d_in, const int* in_sizes, int n_in,
                              void* d_out, int out_size, void* d_ws, size_t ws_size,
                              hipStream_t stream) {
  (void)in_sizes; (void)n_in; (void)out_size; (void)ws_size;
  const float* x     = (const float*)d_in[0];
  const float* Wqkv  = (const float*)d_in[1];
  const float* Wproj = (const float*)d_in[2];
  const float* bproj = (const float*)d_in[3];
  const float* Aq    = (const float*)d_in[4];
  const float* Bq    = (const float*)d_in[5];
  const float* Av    = (const float*)d_in[6];
  const float* Bv    = (const float*)d_in[7];
  float* out = (float*)d_out;

  char* ws = (char*)d_ws;
  size_t off = 0;
  auto take = [&](size_t bytes) {
    char* p = ws + off;
    off += (bytes + 255) & ~(size_t)255;
    return p;
  };
  _Float16* xh  = (_Float16*)take((size_t)NTOK * CDIM * 2);
  _Float16* Wh  = (_Float16*)take((size_t)3 * CDIM * CDIM * 2);
  _Float16* Wph = (_Float16*)take((size_t)CDIM * CDIM * 2);
  _Float16* Qb  = (_Float16*)take((size_t)NTOK * CDIM * 2);
  _Float16* Kb  = (_Float16*)take((size_t)NTOK * CDIM * 2);
  _Float16* Vb  = (_Float16*)take((size_t)NTOK * CDIM * 2);
  _Float16* Pin = (_Float16*)take((size_t)NTOK * CDIM * 2);
  float*    Mq  = (float*)take(64 * 64 * 4);
  float*    Mv  = (float*)take(64 * 64 * 4);

  k_convert<<<(NTOK * CDIM + 255) / 256, 256, 0, stream>>>(x, xh, NTOK * CDIM);
  k_convert<<<(CDIM * CDIM + 255) / 256, 256, 0, stream>>>(Wproj, Wph, CDIM * CDIM);
  k_make_m<<<64, 64, 0, stream>>>(Aq, Bq, Mq);
  k_make_m<<<64, 64, 0, stream>>>(Av, Bv, Mv);
  k_fold_w<<<(3 * CDIM * CDIM + 255) / 256, 256, 0, stream>>>(Wqkv, Mq, Mv, Wh);
  k_gemm_qkv<<<dim3(NTOK / 128, (3 * CDIM) / 128), 256, 0, stream>>>(xh, Wh, Qb, Kb, Vb);
  k_flash<<<dim3(NBATCH * NHEADS, NSEQ / 64), 128, 0, stream>>>(Qb, Kb, Vb, Pin);
  k_gemm_proj<<<dim3(NTOK / 128, CDIM / 128), 256, 0, stream>>>(Pin, Wph, bproj, out);
}
